// Seq2SeqNet_60516089200711
// MI455X (gfx1250) — matrix operand order repaired and checked by reading, not compiled, not run on hardware
//
#include <hip/hip_runtime.h>

// ---------------------------------------------------------------------------
// CDNA5 (gfx1250) wave32 WMMA implementation of the seq2seq reference.
// All GEMMs use v_wmma_f32_16x16x32_f16 (f16 in, f32 accumulate).
// ---------------------------------------------------------------------------

typedef _Float16 h16 __attribute__((ext_vector_type(16)));
typedef _Float16 h8  __attribute__((ext_vector_type(8)));
typedef float    f8  __attribute__((ext_vector_type(8)));

#define DEVINL __device__ __forceinline__

constexpr int T_   = 512;
constexpr int B_   = 32;
constexpr int FIN  = 256;
constexpr int H_   = 1024;
constexpr int FC_  = 4096;
constexpr int OUT_ = 1024;
constexpr int GH   = 4 * H_;          // 4096 gate width

// ---- fragment helpers (layouts per CDNA5 ISA §7.12.2) -----------

// A: 16x32 f16 tile of row-major [.., lda] matrix.
// lanes 0-15: row m0+lane, K chunks {k0..k0+7, k0+16..k0+23}
// lanes 16-31: row m0+lane-16, K chunks {k0+8..k0+15, k0+24..k0+31}
DEVINL h16 load_a_frag(const _Float16* __restrict__ A, int lda, int m0, int k0, int lane) {
    int r = m0 + (lane & 15);
    const _Float16* p = A + (size_t)r * lda + k0 + ((lane & 16) ? 8 : 0);
    h8 a0 = *(const h8*)(p);
    h8 a1 = *(const h8*)(p + 16);
    h16 a;
#pragma unroll
    for (int i = 0; i < 8; i++) { a[i] = a0[i]; a[8 + i] = a1[i]; }
    return a;
}

// B = W^T where W is row-major [Ncols, ldw]; B column n == W row n, so each
// lane loads K chunks matching A: lanes 0-15 -> K={k0..k0+7, k0+16..k0+23} of col0+lane,
// lanes 16-31 -> K={k0+8..k0+15, k0+24..k0+31} of col0+lane-16.
DEVINL h16 load_b_frag(const _Float16* __restrict__ W, int ldw, int col0, int k0, int lane) {
    int c = col0 + (lane & 15);
    const _Float16* p = W + (size_t)c * ldw + k0 + ((lane & 16) ? 8 : 0);
    h8 b0 = *(const h8*)(p);
    h8 b1 = *(const h8*)(p + 16);
    h16 b;
#pragma unroll
    for (int i = 0; i < 8; i++) { b[i] = b0[i]; b[8 + i] = b1[i]; }
    return b;
}

DEVINL f8 wmma16(h16 a, h16 b, f8 c) {
    return __builtin_amdgcn_wmma_f32_16x16x32_f16(
        /*neg_a=*/false, a, /*neg_b=*/false, b,
        /*c_mod=*/(short)0, c, /*reuse_a=*/false, /*reuse_b=*/false);
}

DEVINL f8 zero8() { f8 z; 
#pragma unroll
    for (int i = 0; i < 8; i++) z[i] = 0.0f; return z; }

DEVINL float sigm(float x) { return 1.0f / (1.0f + __expf(-x)); }
DEVINL float tanh_f(float x) {
    x = fminf(fmaxf(x, -15.0f), 15.0f);
    float e = __expf(-2.0f * x);
    return (1.0f - e) / (1.0f + e);
}

// ---------------------------------------------------------------------------
// Fused LSTM step: gates = X@Wih^T + Hprev@Whh^T + bias; cell update; emit
// h (f16 for next step, optional secondary f16 strided view, optional f32
// strided view for outputs) and c (f32 ping-pong).
// One wave -> 16 rows x 16 hidden columns, computing all 4 gate tiles.
// ---------------------------------------------------------------------------
__global__ void __launch_bounds__(256)
lstm_step_kernel(const _Float16* __restrict__ X, int ldx, int Kx,
                 const _Float16* __restrict__ Wih,
                 const _Float16* __restrict__ Hprev,
                 const _Float16* __restrict__ Whh,
                 const float* __restrict__ bias,
                 const float* __restrict__ Cin,
                 float* __restrict__ Cout,
                 _Float16* __restrict__ Hout,
                 _Float16* __restrict__ Hout2, long long ldh2,
                 float* __restrict__ Hout32, long long ldh32,
                 int N)
{
    int tid  = blockIdx.x * blockDim.x + threadIdx.x;
    int wave = tid >> 5;
    int lane = tid & 31;
    int m_tiles = N >> 4;
    int m0 = (wave % m_tiles) << 4;
    int n0 = (wave / m_tiles) << 4;
    if (n0 >= H_) return;                       // wave-uniform guard

    f8 acc[4];
#pragma unroll
    for (int g = 0; g < 4; g++) acc[g] = zero8();

    for (int k = 0; k < Kx; k += 32) {
        h16 a = load_a_frag(X, ldx, m0, k, lane);
#pragma unroll
        for (int g = 0; g < 4; g++) {
            h16 b = load_b_frag(Wih, Kx, g * H_ + n0, k, lane);
            acc[g] = wmma16(a, b, acc[g]);
        }
    }
    for (int k = 0; k < H_; k += 32) {
        h16 a = load_a_frag(Hprev, H_, m0, k, lane);
#pragma unroll
        for (int g = 0; g < 4; g++) {
            h16 b = load_b_frag(Whh, H_, g * H_ + n0, k, lane);
            acc[g] = wmma16(a, b, acc[g]);
        }
    }

    int col = n0 + (lane & 15);
    float bi = bias[0 * H_ + col];
    float bf = bias[1 * H_ + col];
    float bg = bias[2 * H_ + col];
    float bo = bias[3 * H_ + col];
    int rbase = m0 + ((lane & 16) ? 8 : 0);     // C/D fragment row mapping
#pragma unroll
    for (int r = 0; r < 8; r++) {
        int row = rbase + r;
        size_t idx = (size_t)row * H_ + col;
        float iv = sigm(acc[0][r] + bi);
        float fv = sigm(acc[1][r] + bf);
        float gv = tanh_f(acc[2][r] + bg);
        float ov = sigm(acc[3][r] + bo);
        float c  = fv * Cin[idx] + iv * gv;
        float h  = ov * tanh_f(c);
        Cout[idx] = c;
        Hout[idx] = (_Float16)h;
        if (Hout2)  Hout2[(long long)row * ldh2 + col]  = (_Float16)h;
        if (Hout32) Hout32[(long long)row * ldh32 + col] = h;
    }
}

// ---------------------------------------------------------------------------
// FC GEMM: out = act(A) @ W^T + bias.  One wave -> 16 rows x 64 cols
// (4 C tiles), A fragment reused across the 4 WMMAs.
// ---------------------------------------------------------------------------
template <bool RELU, bool OUT16>
__global__ void __launch_bounds__(256)
fc_kernel(const _Float16* __restrict__ A, int K,
          const _Float16* __restrict__ W,
          const float* __restrict__ bias,
          _Float16* __restrict__ out16,
          float* __restrict__ out32,
          int Nc, int M)
{
    int tid  = blockIdx.x * blockDim.x + threadIdx.x;
    int wave = tid >> 5;
    int lane = tid & 31;
    int m_tiles = M >> 4;
    int m0 = (wave % m_tiles) << 4;
    int n0 = (wave / m_tiles) << 6;
    if (n0 >= Nc) return;

    f8 acc[4];
#pragma unroll
    for (int j = 0; j < 4; j++) acc[j] = zero8();

    for (int k = 0; k < K; k += 32) {
        h16 a = load_a_frag(A, K, m0, k, lane);
        if (RELU) {
#pragma unroll
            for (int i = 0; i < 16; i++)
                a[i] = (a[i] > (_Float16)0.0f) ? a[i] : (_Float16)0.0f;
        }
#pragma unroll
        for (int j = 0; j < 4; j++) {
            h16 b = load_b_frag(W, K, n0 + j * 16, k, lane);
            acc[j] = wmma16(a, b, acc[j]);
        }
    }

    int rbase = m0 + ((lane & 16) ? 8 : 0);
#pragma unroll
    for (int j = 0; j < 4; j++) {
        int col = n0 + j * 16 + (lane & 15);
        float bb = bias[col];
#pragma unroll
        for (int r = 0; r < 8; r++) {
            size_t idx = (size_t)(rbase + r) * Nc + col;
            float v = acc[j][r] + bb;
            if (OUT16) out16[idx] = (_Float16)v;
            else       out32[idx] = v;
        }
    }
}

// ---------------------------------------------------------------------------
// small utility kernels
// ---------------------------------------------------------------------------
__global__ void f32_to_f16_kernel(const float* __restrict__ s, _Float16* __restrict__ d, int n) {
    int i = blockIdx.x * blockDim.x + threadIdx.x;
    if (i < n) d[i] = (_Float16)s[i];
}
__global__ void add_f32_kernel(const float* __restrict__ a, const float* __restrict__ b,
                               float* __restrict__ o, int n) {
    int i = blockIdx.x * blockDim.x + threadIdx.x;
    if (i < n) o[i] = a[i] + b[i];
}
__global__ void zero16_kernel(float4* __restrict__ p, int n16) {
    int i = blockIdx.x * blockDim.x + threadIdx.x;
    if (i < n16) p[i] = make_float4(0.f, 0.f, 0.f, 0.f);
}
__global__ void copy_f32_kernel(const float* __restrict__ s, float* __restrict__ d, int n) {
    int i = blockIdx.x * blockDim.x + threadIdx.x;
    if (i < n) d[i] = s[i];
}

// ---------------------------------------------------------------------------
extern "C" void kernel_launch(void* const* d_in, const int* in_sizes, int n_in,
                              void* d_out, int out_size, void* d_ws, size_t ws_size,
                              hipStream_t stream)
{
    (void)in_sizes; (void)n_in; (void)out_size; (void)ws_size;

    const float* in_x = (const float*)d_in[0];
    const float* eWih = (const float*)d_in[1];
    const float* eWhh = (const float*)d_in[2];
    const float* eBih = (const float*)d_in[3];
    const float* eBhh = (const float*)d_in[4];
    const float* dWih = (const float*)d_in[5];
    const float* dWhh = (const float*)d_in[6];
    const float* dBih = (const float*)d_in[7];
    const float* dBhh = (const float*)d_in[8];
    const float* fc1W = (const float*)d_in[9];
    const float* fc1b = (const float*)d_in[10];
    const float* fc2W = (const float*)d_in[11];
    const float* fc2b = (const float*)d_in[12];

    // ---- workspace carving (~110 MB) ----
    char* p = (char*)d_ws;
    auto a16 = [&](size_t n) { _Float16* r = (_Float16*)p; p += n * 2; return r; };
    auto a32 = [&](size_t n) { float* r = (float*)p; p += n * 4; return r; };

    _Float16* X16   = a16((size_t)T_ * B_ * FIN);    // input, f16
    _Float16* WIHE  = a16((size_t)GH * FIN);
    _Float16* WHHE  = a16((size_t)GH * H_);
    _Float16* WIHD  = a16((size_t)GH * H_);
    _Float16* WHHD  = a16((size_t)GH * H_);
    _Float16* FC1W  = a16((size_t)FC_ * H_);
    _Float16* FC2W  = a16((size_t)OUT_ * FC_);
    _Float16* EMB16 = a16((size_t)B_ * T_ * H_);     // embedded [B,T,H], f16
    _Float16* HID16 = a16((size_t)2048 * FC_);       // fc1 chunk scratch
    _Float16* HENC  = a16((size_t)2 * B_ * H_);      // encoder h ping-pong
    _Float16* HDEC  = a16((size_t)2 * T_ * H_);      // decoder h ping-pong
    float*    CENC  = a32((size_t)2 * B_ * H_);
    float*    CDEC  = a32((size_t)2 * T_ * H_);
    float*    BIASE = a32(GH);
    float*    BIASD = a32(GH);

    // ---- output carving ----
    float* ys = (float*)d_out;                                  // [32,512,1024]
    float* hn = ys + (size_t)B_ * T_ * H_;                      // [1,512,1024]
    float* cn = hn + (size_t)T_ * H_;                           // [1,512,1024]
    float* eo = cn + (size_t)T_ * H_;                           // [32,512,1024]

    // ---- one-time (per launch) conversions ----
    auto cvt = [&](const float* s, _Float16* d, size_t n) {
        f32_to_f16_kernel<<<(int)((n + 255) / 256), 256, 0, stream>>>(s, d, (int)n);
    };
    cvt(in_x, X16, (size_t)T_ * B_ * FIN);
    cvt(eWih, WIHE, (size_t)GH * FIN);
    cvt(eWhh, WHHE, (size_t)GH * H_);
    cvt(dWih, WIHD, (size_t)GH * H_);
    cvt(dWhh, WHHD, (size_t)GH * H_);
    cvt(fc1W, FC1W, (size_t)FC_ * H_);
    cvt(fc2W, FC2W, (size_t)OUT_ * FC_);

    add_f32_kernel<<<(GH + 255) / 256, 256, 0, stream>>>(eBih, eBhh, BIASE, GH);
    add_f32_kernel<<<(GH + 255) / 256, 256, 0, stream>>>(dBih, dBhh, BIASD, GH);

    auto zero = [&](void* ptr, size_t bytes) {
        int n16 = (int)(bytes / 16);
        zero16_kernel<<<(n16 + 255) / 256, 256, 0, stream>>>((float4*)ptr, n16);
    };
    zero(HENC, (size_t)B_ * H_ * 2);     // buf0 only (step 0 reads it)
    zero(CENC, (size_t)B_ * H_ * 4);
    zero(HDEC, (size_t)T_ * H_ * 2);
    zero(CDEC, (size_t)T_ * H_ * 4);

    // ---- encoder: 512 sequential steps, batch 32 ----
    // waves = (32/16)*(1024/16) = 128 -> 16 blocks of 256
    for (int t = 0; t < T_; t++) {
        const _Float16* xt  = X16 + (size_t)t * B_ * FIN;
        int pi = t & 1, po = (t + 1) & 1;
        lstm_step_kernel<<<16, 256, 0, stream>>>(
            xt, FIN, FIN, WIHE,
            HENC + (size_t)pi * B_ * H_, WHHE, BIASE,
            CENC + (size_t)pi * B_ * H_, CENC + (size_t)po * B_ * H_,
            HENC + (size_t)po * B_ * H_,
            /*Hout2 = embedded[b][t][:] */ EMB16 + (size_t)t * H_, (long long)T_ * H_,
            /*Hout32*/ nullptr, 0,
            B_);
    }

    // ---- decoder: 32 sequential steps, batch 512 ----
    // waves = (512/16)*(1024/16) = 2048 -> 256 blocks of 256
    for (int s = 0; s < B_; s++) {
        const _Float16* xs = EMB16 + (size_t)s * T_ * H_;
        int pi = s & 1, po = (s + 1) & 1;
        lstm_step_kernel<<<256, 256, 0, stream>>>(
            xs, H_, H_, WIHD,
            HDEC + (size_t)pi * T_ * H_, WHHD, BIASD,
            CDEC + (size_t)pi * T_ * H_, CDEC + (size_t)po * T_ * H_,
            HDEC + (size_t)po * T_ * H_,
            nullptr, 0,
            /*Hout32 = ys[s] */ ys + (size_t)s * T_ * H_, (long long)H_,
            T_);
    }

    // h_n = ys[last step], c_n = final decoder cell state (buf0 after step 31)
    {
        int n = T_ * H_;
        copy_f32_kernel<<<(n + 255) / 256, 256, 0, stream>>>(
            ys + (size_t)(B_ - 1) * T_ * H_, hn, n);
        copy_f32_kernel<<<(n + 255) / 256, 256, 0, stream>>>(CDEC, cn, n);
    }

    // ---- MLP: eo = relu(embedded) @ fc1^T + b1  @ fc2^T + b2 ----
    // chunked over rows so hid scratch stays at 2048x4096 f16 (16 MB)
    constexpr int CH = 2048;
    for (int c = 0; c < (B_ * T_) / CH; c++) {
        const _Float16* Ain = EMB16 + (size_t)c * CH * H_;
        // fc1: waves = (2048/16)*(4096/64) = 8192 -> 1024 blocks
        fc_kernel<true, true><<<1024, 256, 0, stream>>>(
            Ain, H_, FC1W, fc1b, HID16, nullptr, FC_, CH);
        // fc2: waves = (2048/16)*(1024/64) = 2048 -> 256 blocks
        fc_kernel<false, false><<<256, 256, 0, stream>>>(
            HID16, FC_, FC2W, fc2b, nullptr, eo + (size_t)c * CH * OUT_, OUT_, CH);
    }
}
